// Qwen3VLWithExpertModel_25864293056661
// MI455X (gfx1250) — compile-verified, matrix operand order checked by reading, not run on hardware
//
#include <hip/hip_runtime.h>
#include <hip/hip_bf16.h>

typedef __attribute__((ext_vector_type(16))) __bf16 v16bf;
typedef __attribute__((ext_vector_type(8)))  float  v8f;

constexpr int BSZ  = 2;
constexpr int SEG0 = 1024, SEG1 = 256, SEG2 = 64;
constexpr int STOT = 1344;          // 1024+256+64
constexpr int NH   = 16, NKV = 8, HD = 128;
constexpr int HID  = 2048, INTER = 6144;
constexpr float EPSV  = 1e-6f;
constexpr float SCALE = 0.08838834764831845f;   // 128^-0.5
constexpr float NEGV  = -1e9f;

__device__ __forceinline__ v8f wmma_bf16(v16bf a, v16bf b, v8f c) {
    return __builtin_amdgcn_wmma_f32_16x16x32_bf16(false, a, false, b, (short)0, c, false, false);
}

// Generic-pointer -> LDS byte offset (addrspace(3) pointers are 32-bit).
typedef __attribute__((address_space(3))) __bf16 lds_bf16_t;
__device__ __forceinline__ unsigned lds_off(const __bf16* p) {
    return (unsigned)(unsigned long long)(lds_bf16_t*)p;
}

// ---------------------------------------------------------------------------
// Generic GEMM: C[M,N] = A[M,K](bf16) * Bw[N,K]^T(fp32, converted to bf16 in
// LDS staging) (+ Res[M,N] fp32). A-row and C-row index remapping supports the
// segment<->global-sequence scatter/gather:
//   arow(m) = (m / a_seg) * a_full + a_start + (m % a_seg)
// Block: 256 threads = 8 waves; tile 128x128, k-step 32; each wave does a
// 64x32 sub-tile = 8 WMMA accum tiles.
// A tile is staged with CDNA5 async global->LDS DMA (ASYNCcnt), overlapping
// with the fp32->bf16 conversion of the weight tile on the VALU.
// ---------------------------------------------------------------------------
__global__ __launch_bounds__(256)
void gemm_bf16_wmma(const __bf16* __restrict__ A, int a_seg, int a_full, int a_start,
                    const float* __restrict__ Bw, const float* __restrict__ Res,
                    float* __restrict__ C, int c_seg, int c_full, int c_start,
                    int M, int N, int K) {
    __shared__ __align__(64) __bf16 as[128][32];
    __shared__ __align__(64) __bf16 bs[128][32];
    const int tid = threadIdx.x;
    const int m0  = blockIdx.y * 128;
    const int n0  = blockIdx.x * 128;
    const int w   = tid >> 5;
    const int lane = tid & 31;
    const int wr = w >> 2, wc = w & 3;            // 2 x 4 wave grid
    const int lm = lane & 15, lh = lane >> 4;

    v8f acc[4][2];
    for (int i = 0; i < 4; ++i)
        for (int j = 0; j < 2; ++j)
            acc[i][j] = {};

    const int ldr = tid >> 1;                     // 0..127
    const int kp  = (tid & 1) * 16;

    for (int k0 = 0; k0 < K; k0 += 32) {
        // ---- async A tile stage: 32B/thread via 2x global_load_async_to_lds_b128
        {
            int gm   = m0 + ldr;
            int arow = (gm / a_seg) * a_full + a_start + (gm % a_seg);
            const __bf16* gp = A + (size_t)arow * K + k0 + kp;
            unsigned lp = lds_off(&as[ldr][kp]);
            unsigned long long ga = (unsigned long long)gp;
            asm volatile("global_load_async_to_lds_b128 %0, %1, off offset:0"
                         :: "v"(lp), "v"(ga) : "memory");
            asm volatile("global_load_async_to_lds_b128 %0, %1, off offset:16"
                         :: "v"(lp), "v"(ga) : "memory");
        }
        // ---- prefetch next k-step of the streamed fp32 weight tile
        if (k0 + 32 < K)
            __builtin_prefetch(Bw + (size_t)(n0 + ldr) * K + (k0 + 32) + kp, 0, 1);
        // ---- stage B tile (fp32 weights -> bf16 in LDS), overlaps async A DMA
        {
            const float4* src = (const float4*)(Bw + (size_t)(n0 + ldr) * K + k0 + kp);
            float4 f0 = src[0], f1 = src[1], f2 = src[2], f3 = src[3];
            __bf16* dst = &bs[ldr][kp];
            dst[0]  = (__bf16)f0.x; dst[1]  = (__bf16)f0.y; dst[2]  = (__bf16)f0.z; dst[3]  = (__bf16)f0.w;
            dst[4]  = (__bf16)f1.x; dst[5]  = (__bf16)f1.y; dst[6]  = (__bf16)f1.z; dst[7]  = (__bf16)f1.w;
            dst[8]  = (__bf16)f2.x; dst[9]  = (__bf16)f2.y; dst[10] = (__bf16)f2.z; dst[11] = (__bf16)f2.w;
            dst[12] = (__bf16)f3.x; dst[13] = (__bf16)f3.y; dst[14] = (__bf16)f3.z; dst[15] = (__bf16)f3.w;
        }
        asm volatile("s_wait_asynccnt 0x0" ::: "memory");
        __syncthreads();

        v16bf bf0 = *(const v16bf*)(&bs[wc * 32 + lm][lh * 16]);
        v16bf bf1 = *(const v16bf*)(&bs[wc * 32 + 16 + lm][lh * 16]);
        for (int mt = 0; mt < 4; ++mt) {
            v16bf af = *(const v16bf*)(&as[wr * 64 + mt * 16 + lm][lh * 16]);
            acc[mt][0] = wmma_bf16(af, bf0, acc[mt][0]);
            acc[mt][1] = wmma_bf16(af, bf1, acc[mt][1]);
        }
        __syncthreads();
    }

    for (int mt = 0; mt < 4; ++mt)
        for (int nt = 0; nt < 2; ++nt)
            for (int r = 0; r < 8; ++r) {
                int gm = m0 + wr * 64 + mt * 16 + r + 8 * lh;
                int gn = n0 + wc * 32 + nt * 16 + lm;
                float v = acc[mt][nt][r];
                if (Res) v += Res[(size_t)gm * N + gn];
                int crow = (gm / c_seg) * c_full + c_start + (gm % c_seg);
                C[(size_t)crow * N + gn] = v;
            }
}

// ---------------------------------------------------------------------------
// RMSNorm over `cols` fp32, multiply by w, emit bf16 (activations for WMMA).
// One 256-thread block per row.
// ---------------------------------------------------------------------------
__global__ __launch_bounds__(256)
void rmsnorm_bf16(const float* __restrict__ x, const float* __restrict__ w,
                  __bf16* __restrict__ y, int cols) {
    __shared__ float red[8];
    const int row = blockIdx.x;
    const float* xr = x + (size_t)row * cols;
    float ss = 0.f;
    for (int c = threadIdx.x; c < cols; c += 256) { float v = xr[c]; ss += v * v; }
    for (int off = 16; off; off >>= 1) ss += __shfl_xor(ss, off, 32);
    if ((threadIdx.x & 31) == 0) red[threadIdx.x >> 5] = ss;
    __syncthreads();
    float tot = 0.f;
    for (int i = 0; i < 8; ++i) tot += red[i];
    float inv = rsqrtf(tot / (float)cols + EPSV);
    for (int c = threadIdx.x; c < cols; c += 256)
        y[(size_t)row * cols + c] = (__bf16)(xr[c] * inv * w[c]);
}

// ---------------------------------------------------------------------------
// Per-head RMSNorm + RoPE, write into [B, Hn, STOT, HD] bf16 cache.
// One 128-thread block per (b, s, h). src: [B, Sseg, Hn, HD] fp32.
// ---------------------------------------------------------------------------
__global__ __launch_bounds__(128)
void norm_rope(const float* __restrict__ src, const float* __restrict__ nw,
               const float* __restrict__ cosb, const float* __restrict__ sinb,
               __bf16* __restrict__ dst, int Sseg, int Hn, int start) {
    int idx = blockIdx.x;
    int h = idx % Hn; idx /= Hn;
    int s = idx % Sseg;
    int b = idx / Sseg;
    int d = threadIdx.x;
    const float* row = src + ((size_t)(b * Sseg + s) * Hn + h) * HD;
    float v = row[d];
    float ss = v * v;
    for (int off = 16; off; off >>= 1) ss += __shfl_xor(ss, off, 32);
    __shared__ float red[4];
    if ((threadIdx.x & 31) == 0) red[threadIdx.x >> 5] = ss;
    __syncthreads();
    float inv = rsqrtf((red[0] + red[1] + red[2] + red[3]) / (float)HD + EPSV);
    float xn = v * inv * nw[d];
    int od = d < 64 ? d + 64 : d - 64;
    float xno = row[od] * inv * nw[od];
    float rot = d < 64 ? -xno : xno;
    int pos = start + s;
    size_t ci = ((size_t)b * STOT + pos) * HD + d;
    float o = xn * cosb[ci] + rot * sinb[ci];
    dst[((size_t)(b * Hn + h) * STOT + pos) * HD + d] = (__bf16)o;
}

// V -> transposed bf16 cache [B, NKV, HD, STOT]
__global__ __launch_bounds__(128)
void v_transpose(const float* __restrict__ src, __bf16* __restrict__ dst,
                 int Sseg, int start) {
    int idx = blockIdx.x;
    int h = idx % NKV; idx /= NKV;
    int s = idx % Sseg;
    int b = idx / Sseg;
    int d = threadIdx.x;
    float v = src[((size_t)(b * Sseg + s) * NKV + h) * HD + d];
    dst[((size_t)(b * NKV + h) * HD + d) * STOT + (start + s)] = (__bf16)v;
}

// ---------------------------------------------------------------------------
// Flash attention, one wave per (q-tile of 16, head, batch).
// Scores: 4x wmma over D=128. Block-causal mask recomputed from positions.
// Online softmax; P restaged via LDS (C-layout -> A-operand layout).
// P x V: wmma with K = 32 keys, B-operand from transposed V cache.
// ---------------------------------------------------------------------------
__device__ __forceinline__ float rowmax16(float v) {
    for (int off = 8; off; off >>= 1) v = fmaxf(v, __shfl_xor(v, off, 16));
    return v;
}
__device__ __forceinline__ float rowsum16(float v) {
    for (int off = 8; off; off >>= 1) v += __shfl_xor(v, off, 16);
    return v;
}
__device__ __forceinline__ int blk_id(int p) { return p < 1024 ? 0 : (p < 1280 ? 1 : 2); }

__global__ __launch_bounds__(32)
void attn_flash(const __bf16* __restrict__ Qc, const __bf16* __restrict__ Kc,
                const __bf16* __restrict__ Vt, __bf16* __restrict__ att) {
    const int q0  = blockIdx.x * 16;
    const int h   = blockIdx.y;
    const int b   = blockIdx.z;
    const int kvh = h >> 1;                       // GQA: H/HKV = 2
    const int lane = threadIdx.x;
    const int lm = lane & 15, lh = lane >> 4;

    // Q fragments for this tile (A-operand): lane = q-row, 4 slices over D.
    const __bf16* qbase = Qc + (((size_t)(b * NH + h)) * STOT + q0) * HD;
    v16bf qa[4];
    for (int ks = 0; ks < 4; ++ks)
        qa[ks] = *(const v16bf*)(qbase + (size_t)lm * HD + ks * 32 + lh * 16);

    float run_max[8], run_sum[8];
    int qblk[8];
    for (int r = 0; r < 8; ++r) {
        run_max[r] = -1e30f;
        run_sum[r] = 0.f;
        qblk[r] = blk_id(q0 + r + 8 * lh);
    }
    v8f acc[8];
    for (int t = 0; t < 8; ++t) acc[t] = {};

    __shared__ __align__(64) __bf16 p_lds[16][32];

    const __bf16* kbase0 = Kc + (((size_t)(b * NKV + kvh)) * STOT) * HD;
    const __bf16* vbase  = Vt + (((size_t)(b * NKV + kvh)) * HD) * STOT;

    for (int kt = 0; kt < STOT; kt += 32) {
        v8f sc[2];
        for (int j = 0; j < 2; ++j) {
            const int k0 = kt + j * 16;
            v8f s = {};
            const __bf16* kb = kbase0 + (size_t)(k0 + lm) * HD;
            for (int ks = 0; ks < 4; ++ks) {
                v16bf kf = *(const v16bf*)(kb + ks * 32 + lh * 16);
                s = wmma_bf16(qa[ks], kf, s);
            }
            const int kblk = blk_id(k0 + lm);     // this lane's key column
            for (int r = 0; r < 8; ++r) {
                float v = s[r] * SCALE;
                if (kblk > qblk[r]) v += NEGV;
                s[r] = v;
            }
            sc[j] = s;
        }
        float fscale[8];
        for (int r = 0; r < 8; ++r) {
            float mx = fmaxf(rowmax16(sc[0][r]), rowmax16(sc[1][r]));
            float nm = fmaxf(run_max[r], mx);
            float f  = __expf(run_max[r] - nm);
            run_max[r] = nm;
            fscale[r] = f;
            float p0 = __expf(sc[0][r] - nm);
            float p1 = __expf(sc[1][r] - nm);
            sc[0][r] = p0; sc[1][r] = p1;
            run_sum[r] = run_sum[r] * f + rowsum16(p0) + rowsum16(p1);
        }
        // C-layout -> LDS (row = r + 8*lh, col = key-in-tile)
        for (int j = 0; j < 2; ++j)
            for (int r = 0; r < 8; ++r)
                p_lds[r + 8 * lh][j * 16 + lm] = (__bf16)sc[j][r];
        for (int t = 0; t < 8; ++t)
            for (int r = 0; r < 8; ++r)
                acc[t][r] *= fscale[r];
        __syncthreads();
        // reload P as A-operand: lane = q-row, 16 contiguous key-slots per half
        v16bf pa = *(const v16bf*)(&p_lds[lm][lh * 16]);
        for (int t = 0; t < 8; ++t) {
            int d = t * 16 + lm;                  // B-operand column = head dim
            v16bf vf = *(const v16bf*)(vbase + (size_t)d * STOT + kt + lh * 16);
            acc[t] = wmma_bf16(pa, vf, acc[t]);
        }
        __syncthreads();
    }

    for (int t = 0; t < 8; ++t) {
        int d = t * 16 + lm;
        for (int r = 0; r < 8; ++r) {
            int qpos = q0 + r + 8 * lh;
            float o = acc[t][r] / run_sum[r];
            att[((size_t)b * STOT + qpos) * (NH * HD) + h * HD + d] = (__bf16)o;
        }
    }
}

// inter = silu(gate) * up, bf16
__global__ __launch_bounds__(256)
void silu_mul(const float* __restrict__ g, const float* __restrict__ u,
              __bf16* __restrict__ o, size_t n) {
    size_t i = (size_t)blockIdx.x * 256 + threadIdx.x;
    if (i < n) {
        float gv = g[i];
        float s = gv / (1.0f + __expf(-gv));
        o[i] = (__bf16)(s * u[i]);
    }
}

// ---------------------------------------------------------------------------
extern "C" void kernel_launch(void* const* d_in, const int* in_sizes, int n_in,
                              void* d_out, int out_size, void* d_ws, size_t ws_size,
                              hipStream_t stream) {
    (void)in_sizes; (void)n_in; (void)out_size; (void)ws_size;
    const float* hs[3]   = {(const float*)d_in[0], (const float*)d_in[1], (const float*)d_in[2]};
    const float* cosb    = (const float*)d_in[4];
    const float* sinb    = (const float*)d_in[5];
    const float* in_ln   = (const float*)d_in[6];
    const float* q_w     = (const float*)d_in[7];
    const float* k_w     = (const float*)d_in[8];
    const float* v_w     = (const float*)d_in[9];
    const float* q_nw    = (const float*)d_in[10];
    const float* k_nw    = (const float*)d_in[11];
    const float* o_w     = (const float*)d_in[12];
    const float* post_ln = (const float*)d_in[13];
    const float* gate_w  = (const float*)d_in[14];
    const float* up_w    = (const float*)d_in[15];
    const float* down_w  = (const float*)d_in[16];
    float* out = (float*)d_out;

    const int segS[3]     = {SEG0, SEG1, SEG2};
    const int segStart[3] = {0, SEG0, SEG0 + SEG1};

    // workspace carve-out (max M = BSZ*SEG0 = 2048 rows, buffers reused per segment)
    char* W = (char*)d_ws;
    size_t off = 0;
    auto carve = [&](size_t bytes) { char* p = W + off; off += (bytes + 255) & ~(size_t)255; return p; };
    __bf16* xn     = (__bf16*)carve((size_t)2048 * HID * 2);
    float*  qbuf   = (float*) carve((size_t)2048 * (NH * HD) * 4);
    float*  kbuf   = (float*) carve((size_t)2048 * (NKV * HD) * 4);
    float*  vbuf   = (float*) carve((size_t)2048 * (NKV * HD) * 4);
    __bf16* Qc     = (__bf16*)carve((size_t)BSZ * NH  * STOT * HD * 2);
    __bf16* Kc     = (__bf16*)carve((size_t)BSZ * NKV * STOT * HD * 2);
    __bf16* Vt     = (__bf16*)carve((size_t)BSZ * NKV * HD * STOT * 2);
    __bf16* att    = (__bf16*)carve((size_t)BSZ * STOT * NH * HD * 2);
    float*  attres = (float*) carve((size_t)2048 * HID * 4);
    __bf16* x2     = (__bf16*)carve((size_t)2048 * HID * 2);
    float*  gbuf   = (float*) carve((size_t)2048 * INTER * 4);
    float*  ubuf   = (float*) carve((size_t)2048 * INTER * 4);
    __bf16* ibuf   = (__bf16*)carve((size_t)2048 * INTER * 2);

    // ---- Phase A: per-segment QKV into global caches ----
    for (int i = 0; i < 3; ++i) {
        const int Ss = segS[i], st = segStart[i];
        const int M = BSZ * Ss;
        rmsnorm_bf16<<<M, 256, 0, stream>>>(hs[i], in_ln + i * HID, xn, HID);
        gemm_bf16_wmma<<<dim3((NH * HD) / 128, M / 128), 256, 0, stream>>>(
            xn, M, M, 0, q_w + (size_t)i * (NH * HD) * HID, nullptr,
            qbuf, M, M, 0, M, NH * HD, HID);
        gemm_bf16_wmma<<<dim3((NKV * HD) / 128, M / 128), 256, 0, stream>>>(
            xn, M, M, 0, k_w + (size_t)i * (NKV * HD) * HID, nullptr,
            kbuf, M, M, 0, M, NKV * HD, HID);
        gemm_bf16_wmma<<<dim3((NKV * HD) / 128, M / 128), 256, 0, stream>>>(
            xn, M, M, 0, v_w + (size_t)i * (NKV * HD) * HID, nullptr,
            vbuf, M, M, 0, M, NKV * HD, HID);
        norm_rope<<<BSZ * Ss * NH, 128, 0, stream>>>(qbuf, q_nw + i * HD, cosb, sinb, Qc, Ss, NH, st);
        norm_rope<<<BSZ * Ss * NKV, 128, 0, stream>>>(kbuf, k_nw + i * HD, cosb, sinb, Kc, Ss, NKV, st);
        v_transpose<<<BSZ * Ss * NKV, 128, 0, stream>>>(vbuf, Vt, Ss, st);
    }

    // ---- Phase B: attention over full sequence ----
    attn_flash<<<dim3(STOT / 16, NH, BSZ), 32, 0, stream>>>(Qc, Kc, Vt, att);

    // ---- Phase C: per-segment O-proj + MLP ----
    for (int i = 0; i < 3; ++i) {
        const int Ss = segS[i], st = segStart[i];
        const int M = BSZ * Ss;
        // attres = att_seg @ o_w^T + hs   (gather att rows from global sequence)
        gemm_bf16_wmma<<<dim3(HID / 128, M / 128), 256, 0, stream>>>(
            att, Ss, STOT, st, o_w + (size_t)i * HID * (NH * HD), hs[i],
            attres, M, M, 0, M, HID, NH * HD);
        rmsnorm_bf16<<<M, 256, 0, stream>>>(attres, post_ln + i * HID, x2, HID);
        gemm_bf16_wmma<<<dim3(INTER / 128, M / 128), 256, 0, stream>>>(
            x2, M, M, 0, gate_w + (size_t)i * INTER * HID, nullptr,
            gbuf, M, M, 0, M, INTER, HID);
        gemm_bf16_wmma<<<dim3(INTER / 128, M / 128), 256, 0, stream>>>(
            x2, M, M, 0, up_w + (size_t)i * INTER * HID, nullptr,
            ubuf, M, M, 0, M, INTER, HID);
        size_t ne = (size_t)M * INTER;
        silu_mul<<<(unsigned)((ne + 255) / 256), 256, 0, stream>>>(gbuf, ubuf, ibuf, ne);
        // out_seg = inter @ down_w^T + attres  (scatter rows into global sequence)
        gemm_bf16_wmma<<<dim3(HID / 128, M / 128), 256, 0, stream>>>(
            ibuf, M, M, 0, down_w + (size_t)i * HID * INTER, attres,
            out, Ss, STOT, st, M, HID, INTER);
    }
}
